// LayerNormMLP_79671643341681
// MI455X (gfx1250) — matrix operand order repaired and checked by reading, not compiled, not run on hardware
//
#include <hip/hip_runtime.h>

typedef __bf16 bf16;
typedef bf16  v16bf __attribute__((ext_vector_type(16)));
typedef bf16  v8bf  __attribute__((ext_vector_type(8)));
typedef float v8f   __attribute__((ext_vector_type(8)));
typedef int   v4i   __attribute__((ext_vector_type(4)));

// ---------------------------------------------------------------------------
// Async global -> LDS copy of one 16B chunk (CDNA5 GLOBAL_LOAD_ASYNC_TO_LDS,
// tracked by ASYNCcnt). Builtin signature (from hipcc diagnostic):
//   (int4 addrspace(1)*, int4 addrspace(3)*, imm offset, imm cpol)
// Falls back to a synchronous copy if the builtin is unavailable.
// ---------------------------------------------------------------------------
__device__ __forceinline__ void async_cp16(const bf16* g, bf16* l) {
#if __has_builtin(__builtin_amdgcn_global_load_async_to_lds_b128)
  __builtin_amdgcn_global_load_async_to_lds_b128(
      (__attribute__((address_space(1))) v4i*)g,
      (__attribute__((address_space(3))) v4i*)l, 0, 0);
#else
  *(v8bf*)l = *(const v8bf*)g;
#endif
}

__device__ __forceinline__ void wait_async_all() {
#if __has_builtin(__builtin_amdgcn_s_wait_asynccnt)
  __builtin_amdgcn_s_wait_asynccnt(0);
#else
  asm volatile("s_wait_asynccnt 0" ::: "memory");
#endif
}

// ---------------------------------------------------------------------------
// LayerNorm over last dim, fp32 in -> bf16 out. One 256-thread block per row
// (D = 2048 -> 8 elems/thread).
// ---------------------------------------------------------------------------
__global__ __launch_bounds__(256)
void ln_fwd_bf16(const float* __restrict__ x, const float* __restrict__ gamma,
                 const float* __restrict__ beta, bf16* __restrict__ out, int D) {
  __shared__ float red1[256];
  __shared__ float red2[256];
  const int t = threadIdx.x;
  const long row = blockIdx.x;
  const float* xr = x + row * (long)D;

  float vals[8];
  float s = 0.f, ss = 0.f;
#pragma unroll
  for (int i = 0; i < 8; ++i) {
    float v = xr[t * 8 + i];
    vals[i] = v;
    s += v;
    ss += v * v;
  }
  red1[t] = s;
  red2[t] = ss;
  __syncthreads();
#pragma unroll
  for (int o = 128; o > 0; o >>= 1) {
    if (t < o) {
      red1[t] += red1[t + o];
      red2[t] += red2[t + o];
    }
    __syncthreads();
  }
  const float inv_d = 1.0f / (float)D;
  const float mean = red1[0] * inv_d;
  const float var  = red2[0] * inv_d - mean * mean;
  const float rstd = rsqrtf(var + 1e-3f);

  bf16* orow = out + row * (long)D;
#pragma unroll
  for (int i = 0; i < 8; ++i) {
    const int c = t * 8 + i;
    float y = (vals[i] - mean) * rstd * gamma[c] + beta[c];
    orow[c] = (bf16)y;
  }
}

// ---------------------------------------------------------------------------
// Transpose + convert: fp32 [R][C] row-major -> bf16 [C][R] row-major.
// One-time bandwidth cost; makes GEMM B-tiles contiguous (N-major), matching
// the WMMA B-fragment layout exactly.
// ---------------------------------------------------------------------------
__global__ __launch_bounds__(256)
void transpose_cvt_bf16(const float* __restrict__ in, bf16* __restrict__ out,
                        int R, int C) {
  __shared__ float tile[32][33];
  const int tx = threadIdx.x, ty = threadIdx.y;
  const long xin = (long)blockIdx.x * 32 + tx;
  const long yin = (long)blockIdx.y * 32 + ty;
#pragma unroll
  for (int i = 0; i < 4; ++i)
    tile[ty + 8 * i][tx] = in[(yin + 8 * i) * (long)C + xin];
  __syncthreads();
  const long xo = (long)blockIdx.y * 32 + tx;
  const long yo = (long)blockIdx.x * 32 + ty;
#pragma unroll
  for (int i = 0; i < 4; ++i)
    out[(yo + 8 * i) * (long)R + xo] = (bf16)tile[tx][ty + 8 * i];
}

// ---------------------------------------------------------------------------
// bf16 WMMA GEMM:  out[M][N] = A[M][K] * Bt[N][K]^T  (+bias, +optional GELU)
//   EPI == 0 : y = gelu_tanh(acc + bias), stored bf16 (activation buffer)
//   EPI == 1 : y = acc + bias, stored fp32 (final output)
// Block = 256 threads (8 waves), block tile 128x128, wave tile 64x32.
// K-tile = 64 (two WMMA K-steps per barrier), double-buffered LDS with
// async global->LDS staging. 16 v_wmma_f32_16x16x32_bf16 per wave per K-tile.
// ---------------------------------------------------------------------------
template <int EPI>
__global__ __launch_bounds__(256)
void gemm_bf16_wmma(const bf16* __restrict__ A, const bf16* __restrict__ Bt,
                    const float* __restrict__ bias, void* __restrict__ outv,
                    int M, int N, int K) {
  __shared__ bf16 ldsA[2][128 * 64];
  __shared__ bf16 ldsB[2][128 * 64];

  const int t    = threadIdx.x;
  const int lane = t & 31;
  const int wave = t >> 5;
  const int half = lane >> 4;   // 16-lane group
  const int lrow = lane & 15;
  const int waveM = wave >> 2;  // 0..1  -> 64-row slab
  const int waveN = wave & 3;   // 0..3  -> 32-col slab

  const long m0 = (long)blockIdx.y * 128;
  const long n0 = (long)blockIdx.x * 128;

  v8f acc[4][2] = {};

  // Cooperative loader: 256 threads x 64B(A) + 64B(B) per K-tile.
  const int lr  = t >> 1;          // 0..127 : tile row
  const int lsg = (t & 1) * 32;    // 0 or 32 : column segment (elements)

  const bf16* gArow = A  + (m0 + lr) * (long)K + lsg;
  const bf16* gBrow = Bt + (n0 + lr) * (long)K + lsg;
  bf16* lAp = &ldsA[0][lr * 64 + lsg];
  bf16* lBp = &ldsB[0][lr * 64 + lsg];
  const int bufStride = 128 * 64;

  auto load_tile = [&](int buf, long k0) {
    const bf16* ga = gArow + k0;
    const bf16* gb = gBrow + k0;
    bf16* la = lAp + buf * bufStride;
    bf16* lb = lBp + buf * bufStride;
#pragma unroll
    for (int c = 0; c < 4; ++c) {
      async_cp16(ga + 8 * c, la + 8 * c);
      async_cp16(gb + 8 * c, lb + 8 * c);
    }
    // warm L2 for the tile after next (gfx1250 global_prefetch_b8 path)
    __builtin_prefetch(ga + 64, 0, 1);
    __builtin_prefetch(gb + 64, 0, 1);
  };

  load_tile(0, 0);
  int cur = 0;

  for (long k0 = 0; k0 < K; k0 += 64) {
    wait_async_all();      // own async loads into ldsX[cur] complete
    __syncthreads();       // => all waves' loads complete; prev compute done
    if (k0 + 64 < K) load_tile(cur ^ 1, k0 + 64);

    const bf16* baseA = &ldsA[cur][0];
    const bf16* baseB = &ldsB[cur][0];
#pragma unroll
    for (int s = 0; s < 2; ++s) {
      // B fragments: lane holds K = 16*half..16*half+15 of column (n%16),
      // contiguous in the N-major LDS tile.
      v16bf bfrag[2];
#pragma unroll
      for (int ni = 0; ni < 2; ++ni) {
        const bf16* pb =
            baseB + (waveN * 32 + ni * 16 + lrow) * 64 + s * 32 + half * 16;
        v8bf blo = *(const v8bf*)pb;
        v8bf bhi = *(const v8bf*)(pb + 8);
        bfrag[ni] = __builtin_shufflevector(blo, bhi, 0, 1, 2, 3, 4, 5, 6, 7,
                                            8, 9, 10, 11, 12, 13, 14, 15);
      }
#pragma unroll
      for (int mi = 0; mi < 4; ++mi) {
        // A fragment: lane holds K = 16*half..16*half+15 of row m%16.
        const bf16* pa =
            baseA + (waveM * 64 + mi * 16 + lrow) * 64 + s * 32 + half * 16;
        v8bf alo = *(const v8bf*)pa;
        v8bf ahi = *(const v8bf*)(pa + 8);
        v16bf afrag = __builtin_shufflevector(alo, ahi, 0, 1, 2, 3, 4, 5, 6, 7,
                                              8, 9, 10, 11, 12, 13, 14, 15);
#pragma unroll
        for (int ni = 0; ni < 2; ++ni) {
          acc[mi][ni] = __builtin_amdgcn_wmma_f32_16x16x32_bf16(
              false, afrag, false, bfrag[ni], (short)0, acc[mi][ni], false,
              false);
        }
      }
    }
    cur ^= 1;
  }

  // Epilogue. C/D layout: lane L, VGPR v -> row = 8*(L/16)+v, col = L%16.
#pragma unroll
  for (int mi = 0; mi < 4; ++mi) {
#pragma unroll
    for (int ni = 0; ni < 2; ++ni) {
      const long col = n0 + waveN * 32 + ni * 16 + lrow;
      const float bc = bias[col];
#pragma unroll
      for (int v = 0; v < 8; ++v) {
        const long row = m0 + waveM * 64 + mi * 16 + 8 * half + v;
        float x = acc[mi][ni][v] + bc;
        if (EPI == 0) {
          // tanh-approx GELU via sigmoid identity: 0.5(1+tanh u) = sigmoid(2u)
          float u2 = 1.5957691216057308f * (x + 0.044715f * x * x * x);
          float g = __fdividef(x, 1.0f + __expf(-u2));
          ((bf16*)outv)[row * (long)N + col] = (bf16)g;
        } else {
          ((float*)outv)[row * (long)N + col] = x;
        }
      }
    }
  }
}

// ---------------------------------------------------------------------------
// Host-side orchestration (graph-capture safe: only kernel launches on stream).
// ---------------------------------------------------------------------------
extern "C" void kernel_launch(void* const* d_in, const int* in_sizes, int n_in,
                              void* d_out, int out_size, void* d_ws,
                              size_t ws_size, hipStream_t stream) {
  const float* inputs = (const float*)d_in[0];
  const float* gamma  = (const float*)d_in[1];
  const float* beta   = (const float*)d_in[2];
  const float* fc1_w  = (const float*)d_in[3];
  const float* fc1_b  = (const float*)d_in[4];
  const float* fc2_w  = (const float*)d_in[5];
  const float* fc2_b  = (const float*)d_in[6];

  const int D = in_sizes[1];            // 2048
  const int F = in_sizes[4];            // 8192
  const int N = in_sizes[0] / D;        // 8192

  char* ws = (char*)d_ws;
  size_t off = 0;
  bf16* ln  = (bf16*)(ws + off); off += (size_t)N * D * sizeof(bf16);
  bf16* w1t = (bf16*)(ws + off); off += (size_t)D * F * sizeof(bf16);
  bf16* act = (bf16*)(ws + off); off += (size_t)N * F * sizeof(bf16);
  bf16* w2t = (bf16*)(ws + off);

  ln_fwd_bf16<<<N, 256, 0, stream>>>(inputs, gamma, beta, ln, D);

  transpose_cvt_bf16<<<dim3(F / 32, D / 32), dim3(32, 8), 0, stream>>>(
      fc1_w, w1t, D, F);
  transpose_cvt_bf16<<<dim3(D / 32, F / 32), dim3(32, 8), 0, stream>>>(
      fc2_w, w2t, F, D);

  gemm_bf16_wmma<0><<<dim3(F / 128, N / 128), 256, 0, stream>>>(
      ln, w1t, fc1_b, (void*)act, N, F, D);

  gemm_bf16_wmma<1><<<dim3(D / 128, N / 128), 256, 0, stream>>>(
      act, w2t, fc2_b, d_out, N, D, F);
}
